// AutoLabelClusterCrossEntropyLoss_76459007803419
// MI455X (gfx1250) — matrix operand order repaired and checked by reading, not compiled, not run on hardware
//
#include <hip/hip_runtime.h>
#include <stdint.h>

typedef __attribute__((ext_vector_type(16))) __bf16 v16bf;
typedef __attribute__((ext_vector_type(8)))  float  v8f;
typedef __attribute__((ext_vector_type(4)))  float  v4f;

#define N_ROWS   262144
#define K_CENT   1024
#define D_DIM    128
#define TILES_K  64      // 1024 / 16 column tiles
#define NBLOCKS  2048    // 262144 / 128 rows per block

// d_ws layout (bytes)
#define WS_PARTIAL_OFF 0        // float[2048]        (8192 B)
#define WS_C2_OFF      8192     // float[1024]        (4096 B)
#define WS_PACKHI_OFF  12288    // uint16[131072]     (262144 B)
#define WS_PACKLO_OFF  274432   // uint16[131072]     (262144 B)
// total: 536576 B

#if defined(__has_builtin) && __has_builtin(__builtin_amdgcn_sqrtf)
#define FAST_SQRT(x) __builtin_amdgcn_sqrtf(x)
#else
#define FAST_SQRT(x) sqrtf(x)
#endif

__device__ __forceinline__ uint16_t bf16hi_rn(float f) {
    uint32_t u = __builtin_bit_cast(uint32_t, f);
    uint32_t r = u + 0x7FFFu + ((u >> 16) & 1u);   // round-to-nearest-even
    return (uint16_t)(r >> 16);
}
__device__ __forceinline__ float bf16_to_f32(uint16_t h) {
    uint32_t u = ((uint32_t)h) << 16;
    return __builtin_bit_cast(float, u);
}
__device__ __forceinline__ __bf16 mk_bf16(uint16_t h) {
    return __builtin_bit_cast(__bf16, h);
}

// ---- ||c_k||^2 for all K centroids --------------------------------------
__global__ void centroid_sqnorm_kernel(const float* __restrict__ c, float* __restrict__ c2) {
    int k = blockIdx.x * 256 + threadIdx.x;
    if (k < K_CENT) {
        float s = 0.f;
        #pragma unroll 4
        for (int d = 0; d < D_DIM; ++d) {
            float v = c[k * D_DIM + d];
            s = fmaf(v, v, s);
        }
        c2[k] = s;
    }
}

// ---- split centroids into bf16 hi/lo, pre-swizzled into WMMA B-fragment
//      order: element (tile t, kchunk kc, lane, i) at ((t*4+kc)*32+lane)*16+i
//      holds c[t*16 + (lane&15)][kc*32 + (lane>>4)*16 + i]
__global__ void pack_centroids_kernel(const float* __restrict__ c,
                                      uint16_t* __restrict__ hi,
                                      uint16_t* __restrict__ lo) {
    int idx = blockIdx.x * 256 + threadIdx.x;          // 0 .. 131071
    int n = idx >> 7;                                  // centroid index (N dim)
    int d = idx & 127;                                 // feature index  (K dim)
    float x = c[idx];
    uint16_t h = bf16hi_rn(x);
    uint16_t l = bf16hi_rn(x - bf16_to_f32(h));
    int t = n >> 4, nn = n & 15;
    int kc = d >> 5, kk = d & 31;
    int lane = ((kk >> 4) << 4) + nn;
    int i = kk & 15;
    int off = (((t * 4 + kc) * 32 + lane) << 4) + i;
    hi[off] = h;
    lo[off] = l;
}

// ---- main fused kernel ---------------------------------------------------
__global__ __launch_bounds__(256) void
cluster_xent_kernel(const float* __restrict__ emb,
                    const uint16_t* __restrict__ bhiBuf,
                    const uint16_t* __restrict__ bloBuf,
                    const float* __restrict__ c2,
                    float* __restrict__ out,
                    float* __restrict__ partial) {
    const int lane = threadIdx.x & 31;
    const int wave = threadIdx.x >> 5;
    const int hi16 = lane >> 4;                        // 0 = lower half, 1 = upper half
    const long baseRow = (long)blockIdx.x * 128 + (long)wave * 16;
    const int m = lane & 15;                           // A-fragment row for this lane
    const float* rowPtr = emb + (baseRow + m) * D_DIM;

    // ---- load A (16 rows x 128) into bf16 hi/lo fragments, kept in VGPRs ----
    v16bf ahi[4], alo[4];
    float sq = 0.f;
    #pragma unroll
    for (int kc = 0; kc < 4; ++kc) {
        const int d0 = kc * 32 + (hi16 << 4);          // lane's K-range start in this chunk
        v4f f0 = *(const v4f*)(rowPtr + d0);
        v4f f1 = *(const v4f*)(rowPtr + d0 + 4);
        v4f f2 = *(const v4f*)(rowPtr + d0 + 8);
        v4f f3 = *(const v4f*)(rowPtr + d0 + 12);
        float vals[16];
        #pragma unroll
        for (int j = 0; j < 4; ++j) {
            vals[j]      = f0[j];
            vals[4 + j]  = f1[j];
            vals[8 + j]  = f2[j];
            vals[12 + j] = f3[j];
        }
        #pragma unroll
        for (int i = 0; i < 16; ++i) {
            float x = vals[i];
            sq = fmaf(x, x, sq);
            uint16_t h = bf16hi_rn(x);
            ahi[kc][i] = mk_bf16(h);
            alo[kc][i] = mk_bf16(bf16hi_rn(x - bf16_to_f32(h)));
        }
    }
    // lane L and lane L^16 cover complementary halves of row (L&15)'s features
    sq += __shfl_xor(sq, 16, 32);                      // now full ||e_row||^2, row = lane&15

    // e2 for the rows this lane's C slots map to: row(slot v) = v + 8*hi16
    float e2v[8];
    #pragma unroll
    for (int v = 0; v < 8; ++v)
        e2v[v] = __shfl(sq, v + (hi16 << 3), 32);

    // streaming softmax / argmin state per C slot
    float mind[8], ssum[8];
    int   amin[8];
    #pragma unroll
    for (int v = 0; v < 8; ++v) { mind[v] = 3.0e38f; ssum[v] = 0.f; amin[v] = 0; }

    const int colInTile = lane & 15;

    // process two column tiles per iteration: two independent WMMA accumulate
    // chains keep the matrix pipe full and hide WMMA->VALU hazards
    for (int tp = 0; tp < TILES_K / 2; ++tp) {
        const int tA = 2 * tp, tB = 2 * tp + 1;
        v8f acc0 = {};
        v8f acc1 = {};
        #pragma unroll
        for (int kc = 0; kc < 4; ++kc) {
            const long offA = ((long)(tA * 4 + kc) * 32 + lane) << 4;  // uint16 units
            const long offB = ((long)(tB * 4 + kc) * 32 + lane) << 4;
            v16bf bhA = *(const v16bf*)(bhiBuf + offA);
            v16bf blA = *(const v16bf*)(bloBuf + offA);
            v16bf bhB = *(const v16bf*)(bhiBuf + offB);
            v16bf blB = *(const v16bf*)(bloBuf + offB);
            // fp32-accurate dot via bf16 split: hi*hi + hi*lo + lo*hi
            acc0 = __builtin_amdgcn_wmma_f32_16x16x32_bf16(false, ahi[kc], false, bhA,
                                                           (short)0, acc0, false, false);
            acc1 = __builtin_amdgcn_wmma_f32_16x16x32_bf16(false, ahi[kc], false, bhB,
                                                           (short)0, acc1, false, false);
            acc0 = __builtin_amdgcn_wmma_f32_16x16x32_bf16(false, ahi[kc], false, blA,
                                                           (short)0, acc0, false, false);
            acc1 = __builtin_amdgcn_wmma_f32_16x16x32_bf16(false, ahi[kc], false, blB,
                                                           (short)0, acc1, false, false);
            acc0 = __builtin_amdgcn_wmma_f32_16x16x32_bf16(false, alo[kc], false, bhA,
                                                           (short)0, acc0, false, false);
            acc1 = __builtin_amdgcn_wmma_f32_16x16x32_bf16(false, alo[kc], false, bhB,
                                                           (short)0, acc1, false, false);
        }

        const int colA = tA * 16 + colInTile;
        const int colB = tB * 16 + colInTile;
        const float c2A = c2[colA];
        const float c2B = c2[colB];

        #pragma unroll
        for (int v = 0; v < 8; ++v) {
            // ---- tile A ----
            {
                float d2   = e2v[v] + c2A - 2.0f * acc0[v];
                float dist = FAST_SQRT(fmaxf(d2, 0.0f));
                float old  = mind[v];
                float nm   = fminf(old, dist);
                float e    = __expf(nm - fmaxf(old, dist));   // exp(-|dist-old|)
                bool  nw   = dist < old;
                ssum[v] = nw ? fmaf(ssum[v], e, 1.0f) : (ssum[v] + e);
                amin[v] = nw ? colA : amin[v];
                mind[v] = nm;
            }
            // ---- tile B ----
            {
                float d2   = e2v[v] + c2B - 2.0f * acc1[v];
                float dist = FAST_SQRT(fmaxf(d2, 0.0f));
                float old  = mind[v];
                float nm   = fminf(old, dist);
                float e    = __expf(nm - fmaxf(old, dist));
                bool  nw   = dist < old;
                ssum[v] = nw ? fmaf(ssum[v], e, 1.0f) : (ssum[v] + e);
                amin[v] = nw ? colB : amin[v];
                mind[v] = nm;
            }
        }
    }

    // ---- merge (min, argmin, sum) across the 16 lanes sharing each row ----
    #pragma unroll
    for (int mask = 8; mask >= 1; mask >>= 1) {
        #pragma unroll
        for (int v = 0; v < 8; ++v) {
            float om = __shfl_xor(mind[v], mask, 32);
            float os = __shfl_xor(ssum[v], mask, 32);
            int   oi = __shfl_xor(amin[v], mask, 32);
            float nm = fminf(mind[v], om);
            float ns = ssum[v] * __expf(nm - mind[v]) + os * __expf(nm - om);
            bool take = (om < mind[v]) || (om == mind[v] && oi < amin[v]);
            amin[v] = take ? oi : amin[v];
            mind[v] = nm;
            ssum[v] = ns;
        }
    }

    // per-row loss = lse - target = log(sum exp(minDist - dist)) = log(ssum)
    float localLoss = 0.f;
    #pragma unroll
    for (int v = 0; v < 8; ++v)
        localLoss += __logf(ssum[v]);

    // labels: lane 0 writes rows 0..7, lane 16 writes rows 8..15 of this wave's strip
    if ((lane & 15) == 0) {
        const long row0 = baseRow + (hi16 << 3);
        #pragma unroll
        for (int v = 0; v < 8; ++v)
            out[1 + row0 + v] = (float)amin[v];
    }

    // wave loss = lower-half rows + upper-half rows
    localLoss += __shfl_xor(localLoss, 16, 32);

    __shared__ float wsum[8];
    if (lane == 0) wsum[wave] = localLoss;
    __syncthreads();
    if (threadIdx.x == 0) {
        float s = 0.f;
        #pragma unroll
        for (int w = 0; w < 8; ++w) s += wsum[w];
        partial[blockIdx.x] = s;                       // overwritten each launch: no init needed
    }
}

// ---- deterministic final reduction --------------------------------------
__global__ void finalize_kernel(const float* __restrict__ partial, float* __restrict__ out) {
    __shared__ float sh[256];
    float s = 0.f;
    for (int i = threadIdx.x; i < NBLOCKS; i += 256)
        s += partial[i];
    sh[threadIdx.x] = s;
    __syncthreads();
    for (int step = 128; step > 0; step >>= 1) {
        if (threadIdx.x < step) sh[threadIdx.x] += sh[threadIdx.x + step];
        __syncthreads();
    }
    if (threadIdx.x == 0)
        out[0] = sh[0] / (float)N_ROWS;
}

extern "C" void kernel_launch(void* const* d_in, const int* in_sizes, int n_in,
                              void* d_out, int out_size, void* d_ws, size_t ws_size,
                              hipStream_t stream) {
    (void)in_sizes; (void)n_in; (void)out_size; (void)ws_size;
    const float* emb = (const float*)d_in[0];          // [262144, 128] f32
    const float* cen = (const float*)d_in[1];          // [1024, 128]  f32
    float* out = (float*)d_out;                        // [0]=loss, [1..N]=labels (as float)

    char* ws = (char*)d_ws;
    float*    partial = (float*)(ws + WS_PARTIAL_OFF);
    float*    c2      = (float*)(ws + WS_C2_OFF);
    uint16_t* packHi  = (uint16_t*)(ws + WS_PACKHI_OFF);
    uint16_t* packLo  = (uint16_t*)(ws + WS_PACKLO_OFF);

    centroid_sqnorm_kernel<<<(K_CENT + 255) / 256, 256, 0, stream>>>(cen, c2);
    pack_centroids_kernel<<<(K_CENT * D_DIM) / 256, 256, 0, stream>>>(cen, packHi, packLo);
    cluster_xent_kernel<<<NBLOCKS, 256, 0, stream>>>(emb, packHi, packLo, c2, out, partial);
    finalize_kernel<<<1, 256, 0, stream>>>(partial, out);
}
